// CrossAttention_83528523972835
// MI455X (gfx1250) — matrix operand order repaired and checked by reading, compiled, not run on hardware
//
#include <hip/hip_runtime.h>
#include <hip/hip_bf16.h>
#include <math.h>

// ---------------------------------------------------------------------------
// CrossAttention pipeline for MI455X (gfx1250, wave32, WMMA).
// All GEMMs run through one generic f16-in / f32-acc WMMA kernel with a
// double-buffered GLOBAL_LOAD_ASYNC_TO_LDS pipeline (ASYNCcnt tracked).
// Contract: K % 32 == 0, lda/ldb % 8 == 0, base pointers 16B aligned.
// M/N edges handled by row clamping (masked at store).
// ---------------------------------------------------------------------------

typedef __attribute__((ext_vector_type(16))) _Float16 v16h;
typedef __attribute__((ext_vector_type(8)))  _Float16 v8h;
typedef __attribute__((ext_vector_type(8)))  float    v8f;

union F16Frag { v16h v; v8h h[2]; };

#define BM 128
#define BN 128
#define BK 32
#define LDT 40   // BK + 8 pad -> 80-byte rows, 16B aligned, bank-conflict friendly

__device__ __forceinline__ void async_tile_b128(unsigned lds_addr, int byte_off,
                                                const _Float16* base)
{
    // GVS mode: mem = SGPR64(base) + VGPR32(byte_off); LDS dst from VGPR.
    asm volatile("global_load_async_to_lds_b128 %0, %1, %2"
                 :: "v"(lds_addr), "v"(byte_off), "s"(base)
                 : "memory");
}

// C[M,N](f32, ldc) = A[M,K](f16, lda) * BT[N,K](f16, ldb)^T + bias[N]
// Batched via blockIdx.z with element strides.
__global__ __launch_bounds__(256)
void gemm_f16_wmma(const _Float16* __restrict__ A, const _Float16* __restrict__ BT,
                   const float* __restrict__ bias, float* __restrict__ C,
                   int M, int N, int K, int lda, int ldb, int ldc,
                   long long strideA, long long strideB, long long strideC)
{
    __shared__ __align__(16) _Float16 sA[2 * BM * LDT];
    __shared__ __align__(16) _Float16 sB[2 * BN * LDT];

    const int bz = blockIdx.z;
    A  += (long long)bz * strideA;
    BT += (long long)bz * strideB;
    C  += (long long)bz * strideC;

    const int m0   = blockIdx.y * BM;
    const int n0   = blockIdx.x * BN;
    const int tid  = threadIdx.x;
    const int lane = tid & 31;
    const int wid  = tid >> 5;
    const int wm   = (wid >> 2) * 64;   // 2 waves along M
    const int wn   = (wid & 3) * 32;    // 4 waves along N
    const int lrow  = lane & 15;
    const int khalf = lane >> 4;

    const unsigned ldsA0 = (unsigned)(uintptr_t)(void*)sA;
    const unsigned ldsB0 = (unsigned)(uintptr_t)(void*)sB;

    // This thread's two chunks per tile per matrix (512 chunks / 256 threads).
    const int c0 = tid, c1 = tid + 256;
    const int r0 = c0 >> 2, k80 = (c0 & 3) << 3;
    const int r1 = c1 >> 2, k81 = (c1 & 3) << 3;
    int garA0 = m0 + r0; if (garA0 >= M) garA0 = M - 1;   // clamp: masked at store
    int garA1 = m0 + r1; if (garA1 >= M) garA1 = M - 1;
    int garB0 = n0 + r0; if (garB0 >= N) garB0 = N - 1;
    int garB1 = n0 + r1; if (garB1 >= N) garB1 = N - 1;
    const int aoff0 = garA0 * lda, aoff1 = garA1 * lda;
    const int boff0 = garB0 * ldb, boff1 = garB1 * ldb;
    const unsigned la0 = (r0 * LDT + k80) * 2, la1 = (r1 * LDT + k81) * 2;

    v8f acc[4][2] = {};

    const int Ktiles = K / BK;

    // prologue: fill buffer 0 with K-tile 0
    {
        async_tile_b128(ldsA0 + la0, (aoff0 + k80) * 2, A);
        async_tile_b128(ldsA0 + la1, (aoff1 + k81) * 2, A);
        async_tile_b128(ldsB0 + la0, (boff0 + k80) * 2, BT);
        async_tile_b128(ldsB0 + la1, (boff1 + k81) * 2, BT);
    }

    for (int t = 0; t < Ktiles; ++t) {
        const int cur = t & 1;
        if (t + 1 < Ktiles) {
            // overlap: fill the other buffer with the next K-tile
            const int kk = (t + 1) * BK;
            const unsigned bofs = (unsigned)((cur ^ 1) * BM * LDT * 2);
            async_tile_b128(ldsA0 + bofs + la0, (aoff0 + kk + k80) * 2, A);
            async_tile_b128(ldsA0 + bofs + la1, (aoff1 + kk + k81) * 2, A);
            async_tile_b128(ldsB0 + bofs + la0, (boff0 + kk + k80) * 2, BT);
            async_tile_b128(ldsB0 + bofs + la1, (boff1 + kk + k81) * 2, BT);
            // in-order completion: <=4 outstanding means current buffer landed
            asm volatile("s_wait_asynccnt 0x4" ::: "memory");
        } else {
            asm volatile("s_wait_asynccnt 0x0" ::: "memory");
        }
        __syncthreads();

        const _Float16* sAc = sA + cur * BM * LDT;
        const _Float16* sBc = sB + cur * BN * LDT;

        // ---- fragments per ISA 16-bit layouts -----------------------------
        v16h af[4], bfr[2];
        for (int i = 0; i < 4; ++i) {
            // A 16x32: lane (khalf*16+M); K = kb..kb+7 and kb+16..kb+23, kb = khalf*8
            const _Float16* p = &sAc[(wm + i * 16 + lrow) * LDT + khalf * 8];
            F16Frag f;
            f.h[0] = *(const v8h*)p;
            f.h[1] = *(const v8h*)(p + 16);
            af[i] = f.v;
        }
        for (int j = 0; j < 2; ++j) {
            // B 32x16: same K order as A: K = kb..kb+7 and kb+16..kb+23, kb = khalf*8
            const _Float16* p = &sBc[(wn + j * 16 + lrow) * LDT + khalf * 8];
            F16Frag f;
            f.h[0] = *(const v8h*)p;
            f.h[1] = *(const v8h*)(p + 16);
            bfr[j] = f.v;
        }
        for (int i = 0; i < 4; ++i)
            for (int j = 0; j < 2; ++j)
                acc[i][j] = __builtin_amdgcn_wmma_f32_16x16x32_f16(
                    false, af[i], false, bfr[j], (short)0, acc[i][j], false, false);
        __syncthreads();   // all reads of buf[cur] done before it is refilled
    }

    // ---- store: C/D 16x16 f32 layout (VGPR r -> M = r + khalf*8) ----------
    for (int i = 0; i < 4; ++i) {
        int row = m0 + wm + i * 16 + (khalf << 3);
        for (int j = 0; j < 2; ++j) {
            int col = n0 + wn + j * 16 + lrow;
            if (col < N) {
                float bv = bias ? bias[col] : 0.0f;
                for (int r = 0; r < 8; ++r) {
                    int rr = row + r;
                    if (rr < M) C[(long long)rr * ldc + col] = acc[i][j][r] + bv;
                }
            }
        }
    }
}

// ---------------------------------------------------------------------------
// Elementwise / reduction kernels
// ---------------------------------------------------------------------------

// patchify: x[64,128,56,56] -> A[4096,6272] f16, col = (p1*7+p2)*128+c
__global__ void k_patchify(const float* __restrict__ x, _Float16* __restrict__ A,
                           long long total)
{
    long long i = (long long)blockIdx.x * 256 + threadIdx.x;
    if (i >= total) return;
    int col = (int)(i % 6272);
    long long row = i / 6272;
    int c  = col % 128;
    int pp = col / 128;
    int p1 = pp / 7, p2 = pp % 7;
    int bi = (int)(row / 64);
    int patch = (int)(row % 64);
    int ph = patch / 8, pw = patch % 8;
    long long src = (((long long)bi * 128 + c) * 56 + (ph * 7 + p1)) * 56 + (pw * 7 + p2);
    A[i] = (_Float16)x[src];
}

// W[K,N] f32 -> WT[N,K] f16 (coalesced read)
__global__ void k_transpose16(const float* __restrict__ W, _Float16* __restrict__ WT,
                              int K, int N)
{
    long long i = (long long)blockIdx.x * 256 + threadIdx.x;
    long long total = (long long)K * N;
    if (i >= total) return;
    int n = (int)(i % N);
    long long k = i / N;
    WT[(long long)n * K + k] = (_Float16)W[i];
}

// tokens[64,129,2048] f16 = concat(pos[65,2048], patchGEMM[64*64,2048])
__global__ void k_build_tokens(const float* __restrict__ pc, const float* __restrict__ pos,
                               _Float16* __restrict__ t, long long total)
{
    long long i = (long long)blockIdx.x * 256 + threadIdx.x;
    if (i >= total) return;
    int dim = (int)(i % 2048);
    long long r = i / 2048;
    int n = (int)(r % 129);
    int b = (int)(r / 129);
    float v = (n < 65) ? pos[(long long)n * 2048 + dim]
                       : pc[((long long)b * 64 + (n - 65)) * 2048 + dim];
    t[i] = (_Float16)v;
}

__device__ inline unsigned hash_u32(unsigned x)
{
    x ^= x >> 16; x *= 0x7feb352du;
    x ^= x >> 15; x *= 0x846ca68bu;
    x ^= x >> 16;
    return x;
}

// z = k + exp(0.5 q) * eps, scaled by DH^-0.5, laid out [bh*129+n][256] f16
__global__ void k_reparam(const float* __restrict__ qp, const float* __restrict__ kp,
                          _Float16* __restrict__ zs, long long total)
{
    long long i = (long long)blockIdx.x * 256 + threadIdx.x;
    if (i >= total) return;
    int d = (int)(i % 256);
    long long r = i / 256;
    int n = (int)(r % 129);
    long long r2 = r / 129;
    int h = (int)(r2 % 8);
    int b = (int)(r2 / 8);
    long long src = ((long long)b * 129 + n) * 2048 + h * 256 + d;
    float qv = qp[src], kv = kp[src];
    unsigned u1 = hash_u32((unsigned)(2 * i + 1) ^ 0x9E3779B9u);
    unsigned u2 = hash_u32((unsigned)(2 * i + 2) ^ 0x85EBCA6Bu);
    float f1 = ((float)u1 + 0.5f) * (1.0f / 4294967296.0f);
    float f2 = ((float)u2 + 0.5f) * (1.0f / 4294967296.0f);
    float eps = sqrtf(-2.0f * logf(f1)) * cosf(6.28318530718f * f2);
    float z = kv + expf(0.5f * qv) * eps;
    zs[i] = (_Float16)(z * 0.0625f);   // 256^-0.5
}

// softmax over 129 cols -> f16 padded to 160 (pad zeroed so attn GEMM K=160)
__global__ void k_softmax129(const float* __restrict__ dots, _Float16* __restrict__ attn)
{
    int row = blockIdx.x;
    const float* in = dots + (long long)row * 129;
    __shared__ float red[128];
    int t = threadIdx.x;
    float m = -1e30f;
    for (int c = t; c < 129; c += 128) m = fmaxf(m, in[c]);
    red[t] = m; __syncthreads();
    for (int s = 64; s > 0; s >>= 1) {
        if (t < s) red[t] = fmaxf(red[t], red[t + s]);
        __syncthreads();
    }
    float mx = red[0]; __syncthreads();
    float sum = 0.0f;
    for (int c = t; c < 129; c += 128) sum += expf(in[c] - mx);
    red[t] = sum; __syncthreads();
    for (int s = 64; s > 0; s >>= 1) {
        if (t < s) red[t] += red[t + s];
        __syncthreads();
    }
    float inv = 1.0f / red[0];
    _Float16* o = attn + (long long)row * 160;
    for (int c = t; c < 129; c += 128) o[c] = (_Float16)(expf(in[c] - mx) * inv);
    if (t < 31) o[129 + t] = (_Float16)0.0f;
}

// vT[bh][d][n(pad 160)] f16 from vproj[b*129+n][h*256+d] f32
__global__ void k_pack_vT(const float* __restrict__ vp, _Float16* __restrict__ vT,
                          long long total)
{
    long long i = (long long)blockIdx.x * 256 + threadIdx.x;
    if (i >= total) return;
    int np = (int)(i % 160);
    long long r = i / 160;
    int d = (int)(r % 256);
    long long bh = r / 256;
    int h = (int)(bh % 8);
    int b = (int)(bh / 8);
    _Float16 val = (_Float16)0.0f;
    if (np < 129)
        val = (_Float16)vp[((long long)b * 129 + np) * 2048 + h * 256 + d];
    vT[i] = val;
}

// out_heads[bh][n][d] f32 -> attnout[b*129+n][h*256+d] f16
__global__ void k_reorder_heads(const float* __restrict__ oh, _Float16* __restrict__ ao,
                                long long total)
{
    long long i = (long long)blockIdx.x * 256 + threadIdx.x;
    if (i >= total) return;
    int d = (int)(i % 256);
    long long r = i / 256;
    int n = (int)(r % 129);
    long long bh = r / 129;
    int h = (int)(bh % 8);
    int b = (int)(bh / 8);
    ao[((long long)b * 129 + n) * 2048 + h * 256 + d] = (_Float16)oh[i];
}

// LayerNorm over rows of 2048 -> f16
__global__ void k_layernorm2048(const float* __restrict__ x, const float* __restrict__ gamma,
                                const float* __restrict__ beta, _Float16* __restrict__ y)
{
    int row = blockIdx.x;
    const float* in = x + (long long)row * 2048;
    __shared__ float r1[256], r2[256];
    int t = threadIdx.x;
    float s = 0.0f, ss = 0.0f;
    for (int c = t; c < 2048; c += 256) { float v = in[c]; s += v; ss += v * v; }
    r1[t] = s; r2[t] = ss; __syncthreads();
    for (int st = 128; st > 0; st >>= 1) {
        if (t < st) { r1[t] += r1[t + st]; r2[t] += r2[t + st]; }
        __syncthreads();
    }
    float mean = r1[0] * (1.0f / 2048.0f);
    float var  = r2[0] * (1.0f / 2048.0f) - mean * mean;
    float inv  = rsqrtf(var + 1e-5f);
    _Float16* o = y + (long long)row * 2048;
    for (int c = t; c < 2048; c += 256)
        o[c] = (_Float16)((in[c] - mean) * inv * gamma[c] + beta[c]);
}

// per-batch L2 norms over 129*2048 contiguous floats for q and k
__global__ void k_batch_norms(const float* __restrict__ q, const float* __restrict__ k,
                              float* __restrict__ norms)
{
    int b = blockIdx.x;
    const float* qb = q + (long long)b * 264192;
    const float* kb = k + (long long)b * 264192;
    __shared__ float r1[256], r2[256];
    int t = threadIdx.x;
    float sq = 0.0f, sk = 0.0f;
    for (int i = t; i < 264192; i += 256) {
        float a = qb[i]; sq += a * a;
        float c = kb[i]; sk += c * c;
    }
    r1[t] = sq; r2[t] = sk; __syncthreads();
    for (int s = 128; s > 0; s >>= 1) {
        if (t < s) { r1[t] += r1[t + s]; r2[t] += r2[t + s]; }
        __syncthreads();
    }
    if (t == 0) { norms[b] = sqrtf(r1[0]); norms[64 + b] = sqrtf(r2[0]); }
}

// qn/kn in [b][h][n][d] order, normalized per batch
__global__ void k_write_norm(const float* __restrict__ q, const float* __restrict__ k,
                             const float* __restrict__ norms,
                             float* __restrict__ oq, float* __restrict__ ok_,
                             long long total)
{
    long long i = (long long)blockIdx.x * 256 + threadIdx.x;
    if (i >= total) return;
    int d = (int)(i % 256);
    long long r = i / 256;
    int n = (int)(r % 129);
    long long r2 = r / 129;
    int h = (int)(r2 % 8);
    int b = (int)(r2 / 8);
    long long src = ((long long)b * 129 + n) * 2048 + h * 256 + d;
    oq[i]  = q[src] / norms[b];
    ok_[i] = k[src] / norms[64 + b];
}

// ---------------------------------------------------------------------------
// Host orchestration
// ---------------------------------------------------------------------------

static inline void launch_gemm(const _Float16* A, const _Float16* BT, const float* bias,
                               float* C, int M, int N, int K, int lda, int ldb, int ldc,
                               long long sA, long long sB, long long sC, int batch,
                               hipStream_t st)
{
    dim3 g((N + BN - 1) / BN, (M + BM - 1) / BM, batch), blk(256);
    gemm_f16_wmma<<<g, blk, 0, st>>>(A, BT, bias, C, M, N, K, lda, ldb, ldc, sA, sB, sC);
}

static inline long long g1d(long long total) { return (total + 255) / 256; }

extern "C" void kernel_launch(void* const* d_in, const int* in_sizes, int n_in,
                              void* d_out, int out_size, void* d_ws, size_t ws_size,
                              hipStream_t stream)
{
    (void)in_sizes; (void)n_in; (void)out_size; (void)ws_size;
    const float* x_q   = (const float*)d_in[0];
    const float* x_kv  = (const float*)d_in[1];
    const float* W_ex  = (const float*)d_in[2];
    const float* b_ex  = (const float*)d_in[3];
    const float* W_en  = (const float*)d_in[4];
    const float* b_en  = (const float*)d_in[5];
    const float* pos   = (const float*)d_in[6];
    const float* Wq    = (const float*)d_in[7];
    const float* Wk    = (const float*)d_in[8];
    const float* Wv    = (const float*)d_in[9];
    const float* Wo    = (const float*)d_in[10];
    const float* bo    = (const float*)d_in[11];
    const float* gamma = (const float*)d_in[12];
    const float* beta  = (const float*)d_in[13];
    const float* Wh    = (const float*)d_in[14];
    const float* bh_   = (const float*)d_in[15];
    const float* Wm    = (const float*)d_in[16];
    const float* bm    = (const float*)d_in[17];
    float* out = (float*)d_out;
    char* ws = (char*)d_ws;

    size_t off = 0;
    auto alloc = [&](size_t bytes) {
        size_t o = off;
        off += (bytes + 255) & ~(size_t)255;
        return o;
    };

    _Float16* AQ    = (_Float16*)(ws + alloc(4096ull * 6272 * 2));
    _Float16* AK    = (_Float16*)(ws + alloc(4096ull * 6272 * 2));
    _Float16* WEXT  = (_Float16*)(ws + alloc(2048ull * 6272 * 2));
    _Float16* WENT  = (_Float16*)(ws + alloc(2048ull * 6272 * 2));
    float*    PCQ   = (float*)   (ws + alloc(4096ull * 2048 * 4));
    float*    PCK   = (float*)   (ws + alloc(4096ull * 2048 * 4));
    _Float16* TQ    = (_Float16*)(ws + alloc(8256ull * 2048 * 2));
    _Float16* TK    = (_Float16*)(ws + alloc(8256ull * 2048 * 2));
    _Float16* WQT   = (_Float16*)(ws + alloc(2048ull * 2048 * 2));
    _Float16* WKT   = (_Float16*)(ws + alloc(2048ull * 2048 * 2));
    _Float16* WVT   = (_Float16*)(ws + alloc(2048ull * 2048 * 2));
    _Float16* WOT   = (_Float16*)(ws + alloc(2048ull * 2048 * 2));
    float*    QPROJ = (float*)   (ws + alloc(8256ull * 2048 * 4));
    float*    KPROJ = (float*)   (ws + alloc(8256ull * 2048 * 4));
    float*    VPROJ = (float*)   (ws + alloc(8256ull * 2048 * 4));
    _Float16* ZS    = (_Float16*)(ws + alloc(66048ull * 256 * 2));
    _Float16* WMT   = (_Float16*)(ws + alloc(129ull * 256 * 2));
    float*    DOTS  = (float*)   (ws + alloc(66048ull * 129 * 4));
    _Float16* ATTN  = (_Float16*)(ws + alloc(66048ull * 160 * 2));
    _Float16* VT    = (_Float16*)(ws + alloc(512ull * 256 * 160 * 2));
    float*    OHEAD = (float*)   (ws + alloc(512ull * 129 * 256 * 4));
    _Float16* AOUT  = (_Float16*)(ws + alloc(8256ull * 2048 * 2));
    float*    OUTP  = (float*)   (ws + alloc(8256ull * 2048 * 4));
    _Float16* LN    = (_Float16*)(ws + alloc(8256ull * 2048 * 2));
    _Float16* WHT   = (_Float16*)(ws + alloc(6272ull * 2048 * 2));
    float*    NORMS = (float*)   (ws + alloc(128ull * 4));

    // --- patchify + weight transposes (f32 -> f16) --------------------------
    {
        long long tp = 4096ll * 6272;
        k_patchify<<<g1d(tp), 256, 0, stream>>>(x_q, AQ, tp);
        k_patchify<<<g1d(tp), 256, 0, stream>>>(x_kv, AK, tp);
    }
    k_transpose16<<<g1d(6272ll * 2048), 256, 0, stream>>>(W_ex, WEXT, 6272, 2048);
    k_transpose16<<<g1d(6272ll * 2048), 256, 0, stream>>>(W_en, WENT, 6272, 2048);
    k_transpose16<<<g1d(2048ll * 2048), 256, 0, stream>>>(Wq, WQT, 2048, 2048);
    k_transpose16<<<g1d(2048ll * 2048), 256, 0, stream>>>(Wk, WKT, 2048, 2048);
    k_transpose16<<<g1d(2048ll * 2048), 256, 0, stream>>>(Wv, WVT, 2048, 2048);
    k_transpose16<<<g1d(2048ll * 2048), 256, 0, stream>>>(Wo, WOT, 2048, 2048);
    k_transpose16<<<g1d(256ll * 129), 256, 0, stream>>>(Wm, WMT, 256, 129);
    k_transpose16<<<g1d(2048ll * 6272), 256, 0, stream>>>(Wh, WHT, 2048, 6272);

    // --- patch embedding GEMMs ---------------------------------------------
    launch_gemm(AQ, WEXT, b_ex, PCQ, 4096, 2048, 6272, 6272, 6272, 2048, 0, 0, 0, 1, stream);
    launch_gemm(AK, WENT, b_en, PCK, 4096, 2048, 6272, 6272, 6272, 2048, 0, 0, 0, 1, stream);

    // --- token assembly (pos concat) ---------------------------------------
    k_build_tokens<<<g1d(8256ll * 2048), 256, 0, stream>>>(PCQ, pos, TQ, 8256ll * 2048);
    k_build_tokens<<<g1d(8256ll * 2048), 256, 0, stream>>>(PCK, pos, TK, 8256ll * 2048);

    // --- Q/K/V projections --------------------------------------------------
    launch_gemm(TQ, WQT, nullptr, QPROJ, 8256, 2048, 2048, 2048, 2048, 2048, 0, 0, 0, 1, stream);
    launch_gemm(TK, WKT, nullptr, KPROJ, 8256, 2048, 2048, 2048, 2048, 2048, 0, 0, 0, 1, stream);
    launch_gemm(TK, WVT, nullptr, VPROJ, 8256, 2048, 2048, 2048, 2048, 2048, 0, 0, 0, 1, stream);

    // --- reparameterize + dots GEMM ----------------------------------------
    k_reparam<<<g1d(66048ll * 256), 256, 0, stream>>>(QPROJ, KPROJ, ZS, 66048ll * 256);
    launch_gemm(ZS, WMT, bm, DOTS, 66048, 129, 256, 256, 256, 129, 0, 0, 0, 1, stream);

    // --- softmax + batched attn @ V (K padded 129 -> 160 with zeros) --------
    k_softmax129<<<66048, 128, 0, stream>>>(DOTS, ATTN);
    k_pack_vT<<<g1d(512ll * 256 * 160), 256, 0, stream>>>(VPROJ, VT, 512ll * 256 * 160);
    launch_gemm(ATTN, VT, nullptr, OHEAD, 129, 256, 160, 160, 160, 256,
                129ll * 160, 256ll * 160, 129ll * 256, 512, stream);
    k_reorder_heads<<<g1d(512ll * 129 * 256), 256, 0, stream>>>(OHEAD, AOUT, 512ll * 129 * 256);

    // --- output projection + LayerNorm + head GEMM --------------------------
    launch_gemm(AOUT, WOT, bo, OUTP, 8256, 2048, 2048, 2048, 2048, 2048, 0, 0, 0, 1, stream);
    k_layernorm2048<<<8256, 256, 0, stream>>>(OUTP, gamma, beta, LN);

    float* out_head = out;                       // 8256 * 6272
    float* out_qn   = out + 51781632ll;          // 64 * 264192
    float* out_kn   = out_qn + 16908288ll;
    launch_gemm(LN, WHT, bh_, out_head, 8256, 6272, 2048, 2048, 2048, 6272, 0, 0, 0, 1, stream);

    // --- q/k flattened L2 normalization -------------------------------------
    k_batch_norms<<<64, 256, 0, stream>>>(QPROJ, KPROJ, NORMS);
    k_write_norm<<<g1d(64ll * 264192), 256, 0, stream>>>(QPROJ, KPROJ, NORMS,
                                                         out_qn, out_kn, 64ll * 264192);
}
